// ZoneoutGRU_52682068853014
// MI455X (gfx1250) — matrix operand order repaired and checked by reading, not compiled, not run on hardware
//
#include <hip/hip_runtime.h>

typedef __attribute__((ext_vector_type(16))) __bf16 v16bf;
typedef __attribute__((ext_vector_type(8)))  float  v8f;

#define B_  64
#define T_  1024
#define IN_ 512
#define H_  1024
#define ZO_ 0.1f

__device__ __forceinline__ __bf16 f2b(float f) { return (__bf16)f; }
__device__ __forceinline__ float sigmoidf_(float x) { return 1.0f / (1.0f + __expf(-x)); }

// Async copy 16 bytes global -> LDS (CDNA5 GLOBAL_LOAD_ASYNC_TO_LDS_B128,
// tracked by ASYNCcnt). lds pointer is generic (from __shared__); cast to
// addrspace(3) so the "v" operand is the 32-bit LDS offset VGPR the
// instruction expects; global address goes in a 64-bit VGPR pair.
__device__ __forceinline__ void async_g2l_b128(unsigned* lds_p, const __bf16* g) {
    auto lp = (__attribute__((address_space(3))) unsigned*)lds_p;
    asm volatile("global_load_async_to_lds_b128 %0, %1, off"
                 :: "v"(lp), "v"(g)
                 : "memory");
}

// ---------------------------------------------------------------------------
// Weight pack: W is [3H, K] row-major fp32 (row = gate output column).
// Pack into WMMA B-fragment layout for 16x16x32 bf16:
//   B tile (ntile, ktile): lane l holds column n = ntile*16 + (l&15),
//   K values kb..kb+15 where kb = ktile*32 + (l>=16 ? 16 : 0), 16 contiguous
//   bf16 per lane (one v16bf), stored contiguously per (ntile, ktile, lane).
// ---------------------------------------------------------------------------
__global__ void zgru_pack_w(const float* __restrict__ W, int K,
                            __bf16* __restrict__ outp) {
    int tid = blockIdx.x * 256 + threadIdx.x;
    int nK = K >> 5;
    int total = (3 * H_ / 16) * nK * 32;   // lane-vectors
    if (tid >= total) return;
    int lane  = tid & 31;
    int kt    = (tid >> 5) % nK;
    int ntile = tid / (nK * 32);
    int n  = ntile * 16 + (lane & 15);
    int kb = kt * 32 + ((lane >> 4) << 4);
    const float* src = W + (long)n * K + kb;
    __bf16* dst = outp + (long)tid * 16;
#pragma unroll
    for (int e = 0; e < 16; ++e) dst[e] = f2b(src[e]);
}

__global__ void zgru_cvt_bf16(const float* __restrict__ in,
                              __bf16* __restrict__ outp, long n) {
    long i = (long)blockIdx.x * blockDim.x + threadIdx.x;
    if (i < n) outp[i] = f2b(in[i]);
}

__global__ void zgru_init_state(const float* __restrict__ h0in,
                                float* __restrict__ h0f, float* __restrict__ h1f,
                                __bf16* __restrict__ h0b, __bf16* __restrict__ h1b) {
    int i = blockIdx.x * 256 + threadIdx.x;
    if (i < B_ * H_) {
        float a = h0in[i];
        float c = h0in[B_ * H_ + i];
        h0f[i] = a; h0b[i] = f2b(a);
        h1f[i] = c; h1b[i] = f2b(c);
    }
}

__global__ void zgru_tail(const float* __restrict__ h0f,
                          const float* __restrict__ h1f,
                          float* __restrict__ dst) {
    int i = blockIdx.x * 256 + threadIdx.x;
    if (i < B_ * H_) {
        dst[i] = h0f[i];
        dst[B_ * H_ + i] = h1f[i];
    }
}

// ---------------------------------------------------------------------------
// Inner compute for one staged 64x32 A-chunk: build the A fragment from LDS
// (contiguous K order, matching packed B) and run the three gate WMMAs.
// ---------------------------------------------------------------------------
__device__ __forceinline__ void zgru_chunk_mma(
    const unsigned* __restrict__ buf, const v16bf* __restrict__ pBv,
    long base, long gstride, int w4, int m, int hi,
    v8f& accR, v8f& accZ, v8f& accN)
{
    // A fragment, contiguous K order (matches packed B):
    // lanes 0-15: v0-7 -> K pairs 0-7
    // lanes16-31: v0-7 -> K pairs 8-15
    union { unsigned u[8]; v16bf v; } af;
#pragma unroll
    for (int j = 0; j < 8; ++j) {
        int pair = j + hi * 8;
        af.u[j] = buf[(w4 * 16 + m) * 16 + pair];
    }
    v16bf br = pBv[base];
    v16bf bz = pBv[base + gstride];
    v16bf bn = pBv[base + 2 * gstride];
    accR = __builtin_amdgcn_wmma_f32_16x16x32_bf16(false, af.v, false, br, (short)0, accR, false, false);
    accZ = __builtin_amdgcn_wmma_f32_16x16x32_bf16(false, af.v, false, bz, (short)0, accZ, false, false);
    accN = __builtin_amdgcn_wmma_f32_16x16x32_bf16(false, af.v, false, bn, (short)0, accN, false, false);
}

// ---------------------------------------------------------------------------
// One GEMM phase: acc{R,Z,N} += A[64,K] * W_gate^T for 3 gate column blocks.
// A is bf16 row-major (row stride lda elements). Each wave owns one 16-row
// M-tile (w4) and this block's 16-column N-tile; kg in {0,1} is the split-K
// group (processes kt = 2*i + kg). A chunks (64x32 bf16) are staged with
// double-buffered async global->LDS copies, pipelined one chunk ahead so the
// copy latency overlaps the WMMAs of the previous chunk. The last iteration
// is peeled so the steady-state loop body is branch-free.
// ---------------------------------------------------------------------------
__device__ __forceinline__ void zgru_gemm_phase(
    const __bf16* __restrict__ A, long lda, int K,
    const __bf16* __restrict__ pB, int n0t,
    int tid, int lane, int w4, int kg,
    unsigned* ldsG,                  // this K-group's double buffer [2][64*16]
    v8f& accR, v8f& accZ, v8f& accN)
{
    const int nK = K >> 5;
    const int nI = nK >> 1;                          // iterations for this group
    const long gstride = (long)(H_ / 16) * nK * 32;  // v16bf stride per gate
    const v16bf* pBv = (const v16bf*)pB;
    const int tig  = tid & 127;
    const int row  = tig >> 1;
    const int half = tig & 1;
    const int m  = lane & 15;
    const int hi = lane >> 4;
    const __bf16* gRow = A + (long)row * lda + (half << 4);
    unsigned* lRow = ldsG + row * 16 + half * 8;

    // Prologue: issue async stage of this group's first chunk (kt = kg).
    {
        const __bf16* g = gRow + (kg << 5);
        async_g2l_b128(lRow, g);
        async_g2l_b128(lRow + 4, g + 8);
    }

    // Steady state: always issue the next chunk, wait only for the older one
    // (async loads complete in order -> wait cnt<=2).
    for (int i = 0; i < nI - 1; ++i) {
        const int kt = (i << 1) + kg;
        {
            const __bf16* g = gRow + ((kt + 2) << 5);
            unsigned* l = lRow + ((i + 1) & 1) * (64 * 16);
            async_g2l_b128(l, g);
            async_g2l_b128(l + 4, g + 8);
        }
        asm volatile("s_wait_asynccnt 0x2" ::: "memory");
        __syncthreads();   // all waves' chunk-i data landed in LDS
        zgru_chunk_mma(ldsG + (i & 1) * (64 * 16), pBv,
                       ((long)n0t * nK + kt) * 32 + lane, gstride,
                       w4, m, hi, accR, accZ, accN);
        __syncthreads();   // reads done before this buffer is overwritten
    }

    // Epilogue: final chunk (no further issue).
    {
        const int i = nI - 1;
        const int kt = (i << 1) + kg;
        asm volatile("s_wait_asynccnt 0x0" ::: "memory");
        __syncthreads();
        zgru_chunk_mma(ldsG + (i & 1) * (64 * 16), pBv,
                       ((long)n0t * nK + kt) * 32 + lane, gstride,
                       w4, m, hi, accR, accZ, accN);
        __syncthreads();
    }
}

// ---------------------------------------------------------------------------
// One GRU cell step for one layer. Grid: 64 blocks (16 hidden cols each),
// 256 threads = 8 waves = 4 M-tiles x 2 split-K groups.
// ---------------------------------------------------------------------------
__global__ __launch_bounds__(256) void zgru_step(
    const __bf16* __restrict__ A1, long lda1, int K1, const __bf16* __restrict__ pB1,
    const __bf16* __restrict__ A2,              const __bf16* __restrict__ pB2,
    const float* __restrict__ b_ih, const float* __restrict__ b_hh,
    const float* __restrict__ h_prev_f,
    float* __restrict__ h_next_f, __bf16* __restrict__ h_next_bf,
    float* __restrict__ outp, int t)
{
    __shared__ __align__(16) unsigned ldsA[2][2][64 * 16];  // [kg][buf][chunk]
    __shared__ float redL[4 * 32 * 32];

    const int tid  = threadIdx.x;
    const int lane = tid & 31;
    const int wave = tid >> 5;
    const int w4 = wave & 3;       // M-tile
    const int kg = wave >> 2;      // split-K group (tid 0..127 -> 0, 128..255 -> 1)
    const int n0t = blockIdx.x;    // N-tile index
    const int n0  = n0t * 16;

    v8f aR = {}, aZ = {}, aN1 = {}, aN2 = {};
    // input projection:  gi = inp @ w_ih^T
    zgru_gemm_phase(A1, lda1, K1, pB1, n0t, tid, lane, w4, kg, &ldsA[kg][0][0], aR, aZ, aN1);
    // hidden projection: gh = h @ w_hh^T
    zgru_gemm_phase(A2, H_, H_, pB2, n0t, tid, lane, w4, kg, &ldsA[kg][0][0], aR, aZ, aN2);

    __syncthreads();
    if (kg == 1) {
        float* dst = &redL[(w4 * 32 + lane) * 32];
#pragma unroll
        for (int j = 0; j < 8; ++j) {
            dst[j]      = aR[j];
            dst[8 + j]  = aZ[j];
            dst[16 + j] = aN1[j];
            dst[24 + j] = aN2[j];
        }
    }
    __syncthreads();
    if (kg == 0) {
        const float* src = &redL[(w4 * 32 + lane) * 32];
#pragma unroll
        for (int j = 0; j < 8; ++j) {
            aR[j]  += src[j];
            aZ[j]  += src[8 + j];
            aN1[j] += src[16 + j];
            aN2[j] += src[24 + j];
        }
        const int m  = lane & 15;
        const int hi = lane >> 4;
        const int n  = n0 + m;
        const float br = b_ih[n] + b_hh[n];
        const float bz = b_ih[H_ + n] + b_hh[H_ + n];
        const float bi = b_ih[2 * H_ + n];
        const float bh = b_hh[2 * H_ + n];
#pragma unroll
        for (int v = 0; v < 8; ++v) {
            int b = w4 * 16 + hi * 8 + v;        // C layout: VGPR v -> M = v (+8 hi)
            float r  = sigmoidf_(aR[v] + br);
            float z  = sigmoidf_(aZ[v] + bz);
            float nn = tanhf(aN1[v] + bi + r * (aN2[v] + bh));
            float hp = h_prev_f[b * H_ + n];
            float hnew = (1.0f - z) * nn + z * hp;
            float hz = ZO_ * hp + (1.0f - ZO_) * hnew;   // eval-mode zoneout
            h_next_f[b * H_ + n]  = hz;
            h_next_bf[b * H_ + n] = f2b(hz);
            if (outp) outp[((long)b * T_ + t) * H_ + n] = hz;
        }
    }
}

// ---------------------------------------------------------------------------
extern "C" void kernel_launch(void* const* d_in, const int* in_sizes, int n_in,
                              void* d_out, int out_size, void* d_ws, size_t ws_size,
                              hipStream_t stream) {
    const float* x      = (const float*)d_in[0];
    const float* h0     = (const float*)d_in[1];
    const float* w_ih_0 = (const float*)d_in[2];
    const float* w_hh_0 = (const float*)d_in[3];
    const float* b_ih_0 = (const float*)d_in[4];
    const float* b_hh_0 = (const float*)d_in[5];
    const float* w_ih_1 = (const float*)d_in[6];
    const float* w_hh_1 = (const float*)d_in[7];
    const float* b_ih_1 = (const float*)d_in[8];
    const float* b_hh_1 = (const float*)d_in[9];
    float* out = (float*)d_out;

    (void)in_sizes; (void)n_in; (void)out_size; (void)ws_size;

    const long BH = (long)B_ * H_;

    // Carve workspace (256-byte aligned chunks).
    char* p = (char*)d_ws;
    auto carve = [&](size_t bytes) {
        char* r = p;
        p += (bytes + 255) & ~(size_t)255;
        return r;
    };
    __bf16* pw_ih0 = (__bf16*)carve((size_t)3 * H_ * IN_ * 2);
    __bf16* pw_hh0 = (__bf16*)carve((size_t)3 * H_ * H_ * 2);
    __bf16* pw_ih1 = (__bf16*)carve((size_t)3 * H_ * H_ * 2);
    __bf16* pw_hh1 = (__bf16*)carve((size_t)3 * H_ * H_ * 2);
    __bf16* xbf    = (__bf16*)carve((size_t)B_ * T_ * IN_ * 2);
    float*  h0f    = (float*)carve((size_t)2 * BH * 4);   // ping-pong
    float*  h1f    = (float*)carve((size_t)2 * BH * 4);
    __bf16* h0b    = (__bf16*)carve((size_t)2 * BH * 2);
    __bf16* h1b    = (__bf16*)carve((size_t)2 * BH * 2);

    // 1) Pack weights into WMMA B-fragment layout (bf16).
    {
        int tot_in = (3 * H_ / 16) * (IN_ / 32) * 32;  // 98304
        int tot_h  = (3 * H_ / 16) * (H_ / 32) * 32;   // 196608
        zgru_pack_w<<<(tot_in + 255) / 256, 256, 0, stream>>>(w_ih_0, IN_, pw_ih0);
        zgru_pack_w<<<(tot_h  + 255) / 256, 256, 0, stream>>>(w_hh_0, H_,  pw_hh0);
        zgru_pack_w<<<(tot_h  + 255) / 256, 256, 0, stream>>>(w_ih_1, H_,  pw_ih1);
        zgru_pack_w<<<(tot_h  + 255) / 256, 256, 0, stream>>>(w_hh_1, H_,  pw_hh1);
    }
    // 2) Convert x to bf16 once.
    {
        long n = (long)B_ * T_ * IN_;
        zgru_cvt_bf16<<<(unsigned)((n + 255) / 256), 256, 0, stream>>>(x, xbf, n);
    }
    // 3) Initialize hidden state buffers (fp32 + bf16).
    zgru_init_state<<<(B_ * H_ + 255) / 256, 256, 0, stream>>>(h0, h0f, h1f, h0b, h1b);

    // 4) Sequential recurrence: two step kernels per time step.
    int ping = 0;
    for (int t = 0; t < T_; ++t) {
        int pong = 1 - ping;
        // Layer 0: inp = x_t (bf16, row stride T*IN)
        zgru_step<<<H_ / 16, 256, 0, stream>>>(
            xbf + (long)t * IN_, (long)T_ * IN_, IN_, pw_ih0,
            h0b + (long)ping * BH, pw_hh0,
            b_ih_0, b_hh_0,
            h0f + (long)ping * BH,
            h0f + (long)pong * BH, h0b + (long)pong * BH,
            (float*)nullptr, t);
        // Layer 1: inp = updated layer-0 state (zoneout already applied)
        zgru_step<<<H_ / 16, 256, 0, stream>>>(
            h0b + (long)pong * BH, (long)H_, H_, pw_ih1,
            h1b + (long)ping * BH, pw_hh1,
            b_ih_1, b_hh_1,
            h1f + (long)ping * BH,
            h1f + (long)pong * BH, h1b + (long)pong * BH,
            out, t);
        ping = pong;
    }

    // 5) Final hidden states -> d_out tail ([L,B,H] after [B,T,H]).
    zgru_tail<<<(B_ * H_ + 255) / 256, 256, 0, stream>>>(
        h0f + (long)ping * BH, h1f + (long)ping * BH, out + (long)B_ * T_ * H_);
}
